// SSTNetwork_62491774157544
// MI455X (gfx1250) — matrix operand order repaired and checked by reading, not compiled, not run on hardware
//
#include <hip/hip_runtime.h>

// ---------------------------------------------------------------------------
// CDNA5 (gfx1250) transformer block: LN -> +PE -> windowed MHA -> LN -> FFN
// Matmuls: v_wmma_f32_16x16x32_f16. Weight panels staged into LDS via the
// Tensor Data Mover (tensor_load_to_lds + s_wait_tensorcnt).
// ---------------------------------------------------------------------------

#ifndef USE_TDM
#define USE_TDM 1
#endif

typedef __attribute__((ext_vector_type(16))) _Float16 v16h;
typedef __attribute__((ext_vector_type(8)))  float    v8f;
typedef __attribute__((ext_vector_type(4)))  unsigned u32x4;
typedef __attribute__((ext_vector_type(8)))  int      i32x8;
typedef __attribute__((ext_vector_type(4)))  int      i32x4;

union AFrag { v16h v; _Float16 h[16]; unsigned u[8]; };
union CFrag { v8f  v; float f[8]; };

// 16-bit A-matrix 16x32 per-lane K order used here (contiguous, not ISA table order):
// lanes 0-15: K = {0..15}; lanes 16-31: K = {16..31}
__device__ __forceinline__ int kpatA(int vv) { return 2 * vv; }
__device__ __forceinline__ int iclamp(int x, int lo, int hi) {
    return x < lo ? lo : (x > hi ? hi : x);
}

#if USE_TDM
// ---------------------------------------------------------------------------
// TDM: DMA a 2D tile (tile_y rows x tile_x contiguous elements, 2B each) from
// global into LDS (row-major, contiguous). D# bitfields per ISA 08 sect 8.
// ---------------------------------------------------------------------------
__device__ __forceinline__ void tdm_load_2d_f16(unsigned lds_byte_off,
                                                const void* gaddr,
                                                unsigned tile_x, unsigned tile_y,
                                                unsigned tensor_x, unsigned tensor_y,
                                                unsigned stride_x_elems) {
    unsigned long long ga = (unsigned long long)gaddr;
    u32x4 g0;
    g0[0] = 1u;                                   // count=1 (valid user descriptor)
    g0[1] = lds_byte_off;                         // lds_addr [63:32]
    g0[2] = (unsigned)(ga & 0xffffffffu);         // global_addr lo
    g0[3] = (unsigned)((ga >> 32) & 0x01ffffffu)  // global_addr [56:32]
            | (2u << 30);                         // type=2 ("image")
    i32x8 g1;
    g1[0] = (int)(1u << 16);                      // data_size=1 -> 2 bytes
    g1[1] = (int)((tensor_x & 0xffffu) << 16);    // tensor_dim0[15:0] @ bits 63:48
    g1[2] = (int)((tensor_x >> 16) | ((tensor_y & 0xffffu) << 16));
    g1[3] = (int)((tensor_y >> 16) | ((tile_x & 0xffffu) << 16));  // tile_dim0
    g1[4] = (int)(tile_y & 0xffffu);              // tile_dim1 (tile_dim2 = 0)
    g1[5] = (int)stride_x_elems;                  // tensor_dim0_stride lo32
    g1[6] = 0;                                    // stride hi16 | dim1_stride lo16
    g1[7] = 0;
    i32x4 gz = {0, 0, 0, 0};
#if __has_include(<hip/amd_detail/amd_gfx1250_TDM.h>)
    i32x8 gz8 = {0, 0, 0, 0, 0, 0, 0, 0};
    __builtin_amdgcn_tensor_load_to_lds(g0, g1, gz, gz, gz8, 0);
#else
    __builtin_amdgcn_tensor_load_to_lds(g0, g1, gz, gz, 0);
#endif
}
#endif  // USE_TDM

// ---------------------------------------------------------------------------
// f32 -> f16 weight conversion
// ---------------------------------------------------------------------------
__global__ __launch_bounds__(256) void cvt_f16(const float* __restrict__ s,
                                               _Float16* __restrict__ d, int n) {
    int i = blockIdx.x * 256 + threadIdx.x;
    if (i < n) d[i] = (_Float16)s[i];
}

// ---------------------------------------------------------------------------
// LayerNorm (+ optional sinusoidal PE + transpose to (B*K, T, D) layout)
// One wave32 per token (8 floats/lane over D=256).
// ---------------------------------------------------------------------------
template <bool ADD_PE>
__global__ __launch_bounds__(256) void ln_kernel(const float* __restrict__ x,
                                                 const float* __restrict__ gain,
                                                 const float* __restrict__ beta,
                                                 _Float16* __restrict__ out) {
    const int tid = threadIdx.x, lane = tid & 31, wave = tid >> 5;
    const int tok = blockIdx.x * 8 + wave;  // flat (b, t, kk)
    const float* xp = x + (size_t)tok * 256 + lane * 8;
    float4 a0 = *reinterpret_cast<const float4*>(xp);
    float4 a1 = *reinterpret_cast<const float4*>(xp + 4);
    float xv[8] = {a0.x, a0.y, a0.z, a0.w, a1.x, a1.y, a1.z, a1.w};

    float s = 0.f;
#pragma unroll
    for (int j = 0; j < 8; j++) s += xv[j];
#pragma unroll
    for (int m = 16; m >= 1; m >>= 1) s += __shfl_xor(s, m, 32);
    const float mu = s * (1.f / 256.f);

    float vs = 0.f;
#pragma unroll
    for (int j = 0; j < 8; j++) { float d = xv[j] - mu; vs += d * d; }
#pragma unroll
    for (int m = 16; m >= 1; m >>= 1) vs += __shfl_xor(vs, m, 32);
    const float inv = rsqrtf(vs * (1.f / 256.f) + 1e-5f);

    const int dbase = lane * 8;
    const int t = (tok >> 2) & 1023;
    float pe[8];
    if (ADD_PE) {
#pragma unroll
        for (int p = 0; p < 4; p++) {
            int d0 = dbase + 2 * p;
            float fr = __expf(-(float)d0 * 0.03597789249f);  // ln(10000)/256
            float sn, cs;
            __sincosf((float)t * fr, &sn, &cs);
            pe[2 * p] = sn;
            pe[2 * p + 1] = cs;
        }
    }

    union { _Float16 h[8]; uint4 u; } ov;
#pragma unroll
    for (int j = 0; j < 8; j++) {
        float val = (xv[j] - mu) * inv * gain[dbase + j] + beta[dbase + j];
        if (ADD_PE) val += pe[j];
        ov.h[j] = (_Float16)val;
    }

    size_t orow;
    if (ADD_PE) {
        int bI = tok >> 12, kk = tok & 3;  // transpose to (b*K+kk, t)
        orow = ((size_t)(bI * 4 + kk)) * 1024 + t;
    } else {
        orow = tok;
    }
    *reinterpret_cast<uint4*>(out + orow * 256 + dbase) = ov.u;
}

// ---------------------------------------------------------------------------
// WMMA GEMM:  Y[8192 x NOUT] = X[8192 x KDIM] * W[NOUT x KDIM]^T + bias
// Block tile 64x64, BK=64; 8 waves, each wave a 16x32 output (2 accumulators)
// -> 4 v_wmma per barrier pair. Weight tile DMA'd into LDS by the TDM.
// MODE 0: f16 store   1: ReLU + f16 store
//      2: f32 = acc + bias + x (permuted back to (B,T,K,D)) -> xres
//      3: f32 = acc + bias + resid (same layout) -> Y
// ---------------------------------------------------------------------------
template <int MODE, int KDIM, int NOUT>
__global__ __launch_bounds__(256) void gemm_tile(const _Float16* __restrict__ X,
                                                 const _Float16* __restrict__ W,
                                                 const float* __restrict__ bias,
                                                 void* __restrict__ Y,
                                                 const float* __restrict__ resid) {
    constexpr int BM = 64, BN = 64, BK = 64;
    __shared__ _Float16 As[BM * BK];  // [row][k], stride BK
    __shared__ _Float16 Bs[BN * BK];  // [n][k],  stride BK
    const int tid = threadIdx.x, lane = tid & 31, wave = tid >> 5;
    const int m0 = blockIdx.x * BM, n0 = blockIdx.y * BN;
    const int mt = (wave & 3) * 16, nt = (wave >> 2) * 32;

    CFrag c[2];
#pragma unroll
    for (int n2 = 0; n2 < 2; n2++)
#pragma unroll
        for (int r = 0; r < 8; r++) c[n2].f[r] = 0.f;

    const int arow = lane & 15, akb = (lane & 16) ? 16 : 0;
    const int bn = lane & 15, bkb = (lane & 16) ? 16 : 0;

    for (int kc = 0; kc < KDIM; kc += BK) {
#if USE_TDM
        if (wave == 0) {
            tdm_load_2d_f16(
                (unsigned)(unsigned long long)(size_t)&Bs[0],
                (const char*)W + ((size_t)n0 * KDIM + kc) * 2,
                /*tile_x=*/BK, /*tile_y=*/BN,
                /*tensor_x=*/KDIM, /*tensor_y=*/NOUT,
                /*stride_x=*/KDIM);
        }
#else
        {  // cooperative B tile: 64x64 halves, 32B per thread
            int r = tid >> 2, o = (tid & 3) * 16;
            const _Float16* src = W + (size_t)(n0 + r) * KDIM + kc + o;
            *reinterpret_cast<uint4*>(&Bs[r * BK + o]) =
                *reinterpret_cast<const uint4*>(src);
            *reinterpret_cast<uint4*>(&Bs[r * BK + o + 8]) =
                *reinterpret_cast<const uint4*>(src + 8);
        }
#endif
        {  // cooperative A tile: 64x64 halves, 32B per thread
            int r = tid >> 2, o = (tid & 3) * 16;
            const _Float16* src = X + (size_t)(m0 + r) * KDIM + kc + o;
            *reinterpret_cast<uint4*>(&As[r * BK + o]) =
                *reinterpret_cast<const uint4*>(src);
            *reinterpret_cast<uint4*>(&As[r * BK + o + 8]) =
                *reinterpret_cast<const uint4*>(src + 8);
            if (kc + BK < KDIM) __builtin_prefetch(src + BK, 0, 1);
        }
#if USE_TDM
        if (wave == 0) __builtin_amdgcn_s_wait_tensorcnt(0);
#endif
        __syncthreads();

#pragma unroll
        for (int kk = 0; kk < 2; kk++) {  // two 32-wide K subchunks
            AFrag a;
#pragma unroll
            for (int vv = 0; vv < 8; vv++)
                a.u[vv] = *reinterpret_cast<const unsigned*>(
                    &As[(mt + arow) * BK + kk * 32 + akb + kpatA(vv)]);
#pragma unroll
            for (int n2 = 0; n2 < 2; n2++) {
                AFrag b;
#pragma unroll
                for (int vv = 0; vv < 8; vv++)
                    b.u[vv] = *reinterpret_cast<const unsigned*>(
                        &Bs[(nt + n2 * 16 + bn) * BK + kk * 32 + bkb + 2 * vv]);
                c[n2].v = __builtin_amdgcn_wmma_f32_16x16x32_f16(
                    false, a.v, false, b.v, (short)0, c[n2].v, false, false);
            }
        }
        __syncthreads();
    }

    const int mb = m0 + mt + ((lane & 16) ? 8 : 0);
#pragma unroll
    for (int n2 = 0; n2 < 2; n2++) {
        const int ncol = n0 + nt + n2 * 16 + (lane & 15);
        const float bv = bias[ncol];
#pragma unroll
        for (int r = 0; r < 8; r++) {
            int row = mb + r;
            float val = c[n2].f[r] + bv;
            if (MODE == 0) {
                reinterpret_cast<_Float16*>(Y)[(size_t)row * NOUT + ncol] = (_Float16)val;
            } else if (MODE == 1) {
                val = fmaxf(val, 0.f);
                reinterpret_cast<_Float16*>(Y)[(size_t)row * NOUT + ncol] = (_Float16)val;
            } else if (MODE == 2) {
                // row is (b*K+kk)*1024 + t ; map back to (b,t,kk,d)
                int nsq = row >> 10, t = row & 1023, bI = nsq >> 2, kk = nsq & 3;
                size_t idx = ((size_t)(bI * 4096 + t * 4 + kk)) * 256 + ncol;
                reinterpret_cast<float*>(Y)[idx] = val + resid[idx];
            } else {
                size_t idx = (size_t)row * 256 + ncol;
                reinterpret_cast<float*>(Y)[idx] = val + resid[idx];
            }
        }
    }
}

// ---------------------------------------------------------------------------
// Windowed attention. One wave per (seq, head, 16-query block).
// DH=32 -> one WMMA per 16x16 score tile; 3 tiles cover the 48-key span.
// Masked softmax -> f16 probs (16x64 zero-padded) -> P x V via 4 WMMAs.
// ---------------------------------------------------------------------------
__global__ __launch_bounds__(256) void attn_kernel(const _Float16* __restrict__ q,
                                                   const _Float16* __restrict__ k,
                                                   const _Float16* __restrict__ v,
                                                   _Float16* __restrict__ o) {
    __shared__ float    S[8][16 * 48];  // per-wave score scratch
    __shared__ _Float16 P[8][16 * 64];  // per-wave prob matrix (padded)
    const int tid = threadIdx.x, lane = tid & 31, wave = tid >> 5;
    const int task = blockIdx.x * 8 + wave;
    const int qb = task & 63, head = (task >> 6) & 7, seq = task >> 9;
    const int t0 = qb * 16;
    const size_t base = ((size_t)seq * 1024) * 256 + head * 32;

    const int arow = lane & 15, akb = (lane & 16) ? 16 : 0;
    const int cn = lane & 15, bkb = (lane & 16) ? 16 : 0;

    AFrag aq;  // Q fragment: 16 queries x 32(DH), K pairs contiguous in d
#pragma unroll
    for (int vv = 0; vv < 8; vv++)
        aq.u[vv] = *reinterpret_cast<const unsigned*>(
            q + base + (size_t)(t0 + arow) * 256 + akb + kpatA(vv));

    const float scale = 0.17677669529663689f;  // 1/sqrt(32)
#pragma unroll
    for (int tt = 0; tt < 3; tt++) {
        const int key = t0 - 16 + tt * 16 + cn;
        const int kcl = iclamp(key, 0, 1023);
        AFrag bk;  // B = K^T tile: [d][key], d pairs contiguous in memory
#pragma unroll
        for (int vv = 0; vv < 8; vv++)
            bk.u[vv] = *reinterpret_cast<const unsigned*>(
                k + base + (size_t)kcl * 256 + bkb + 2 * vv);
        CFrag s;
#pragma unroll
        for (int r = 0; r < 8; r++) s.f[r] = 0.f;
        s.v = __builtin_amdgcn_wmma_f32_16x16x32_f16(false, aq.v, false, bk.v,
                                                     (short)0, s.v, false, false);
#pragma unroll
        for (int r = 0; r < 8; r++) {
            int m = r + ((lane & 16) ? 8 : 0);
            int t = t0 + m;
            bool valid = (key >= 0) && (key < 1024) && (key >= t - 16) && (key <= t + 16);
            S[wave][m * 48 + tt * 16 + cn] = valid ? s.f[r] * scale : -1e30f;
        }
    }

    // softmax over the 48-key span (lanes l, l+16 redundantly own row l&15)
    const float* sp = &S[wave][arow * 48];
    float mx = -1e30f;
    for (int j = 0; j < 48; j++) mx = fmaxf(mx, sp[j]);
    float sum = 0.f;
    for (int j = 0; j < 48; j++) sum += __expf(sp[j] - mx);
    const float rinv = 1.f / sum;  // self position always valid -> sum > 0
    const int jbase = (lane & 16) ? 32 : 0;
    for (int jj = 0; jj < 32; jj++) {
        int j = jbase + jj;
        float pv = (j < 48) ? __expf(sp[j] - mx) * rinv : 0.f;  // zero pad 48..63
        P[wave][arow * 64 + j] = (_Float16)pv;
    }

    // out tile = P(16x64) x V(64x32): 2 col tiles x 2 K chunks
#pragma unroll
    for (int ntile = 0; ntile < 2; ntile++) {
        CFrag acc;
#pragma unroll
        for (int r = 0; r < 8; r++) acc.f[r] = 0.f;
#pragma unroll
        for (int kc2 = 0; kc2 < 2; kc2++) {
            AFrag ap;
#pragma unroll
            for (int vv = 0; vv < 8; vv++)
                ap.u[vv] = *reinterpret_cast<const unsigned*>(
                    &P[wave][arow * 64 + kc2 * 32 + akb + kpatA(vv)]);
            AFrag bvf;  // B[k][n] = V[key(k)][d]: rows strided -> gather halves
#pragma unroll
            for (int vv = 0; vv < 8; vv++) {
                int k2 = kc2 * 32 + bkb + 2 * vv;
                int c0 = iclamp(t0 - 16 + k2, 0, 1023);
                int c1 = iclamp(t0 - 15 + k2, 0, 1023);
                bvf.h[2 * vv]     = v[base + (size_t)c0 * 256 + ntile * 16 + cn];
                bvf.h[2 * vv + 1] = v[base + (size_t)c1 * 256 + ntile * 16 + cn];
            }
            acc.v = __builtin_amdgcn_wmma_f32_16x16x32_f16(false, ap.v, false, bvf.v,
                                                           (short)0, acc.v, false, false);
        }
#pragma unroll
        for (int r = 0; r < 8; r++) {
            int m = r + ((lane & 16) ? 8 : 0);
            o[base + (size_t)(t0 + m) * 256 + ntile * 16 + cn] = (_Float16)acc.f[r];
        }
    }
}

// ---------------------------------------------------------------------------
// Host orchestration
// ---------------------------------------------------------------------------
extern "C" void kernel_launch(void* const* d_in, const int* in_sizes, int n_in,
                              void* d_out, int out_size, void* d_ws, size_t ws_size,
                              hipStream_t stream) {
    (void)in_sizes; (void)n_in; (void)out_size; (void)ws_size;
    const float* x      = (const float*)d_in[0];
    const float* ln_t_g = (const float*)d_in[1];
    const float* ln_t_b = (const float*)d_in[2];
    const float* Wq     = (const float*)d_in[3];
    const float* Wk     = (const float*)d_in[4];
    const float* Wv     = (const float*)d_in[5];
    const float* bq     = (const float*)d_in[6];
    const float* bk     = (const float*)d_in[7];
    const float* bv     = (const float*)d_in[8];
    const float* Wo     = (const float*)d_in[9];
    const float* bo     = (const float*)d_in[10];
    const float* ln_f_g = (const float*)d_in[11];
    const float* ln_f_b = (const float*)d_in[12];
    const float* W1     = (const float*)d_in[13];
    const float* b1     = (const float*)d_in[14];
    const float* W2     = (const float*)d_in[15];
    const float* b2     = (const float*)d_in[16];

    // workspace carve-up (half elements)
    _Float16* hb = (_Float16*)d_ws;
    size_t off = 0;
    auto alloc = [&](size_t n) { _Float16* p = hb + off; off += n; return p; };
    _Float16* wq16 = alloc(256 * 256);
    _Float16* wk16 = alloc(256 * 256);
    _Float16* wv16 = alloc(256 * 256);
    _Float16* wo16 = alloc(256 * 256);
    _Float16* w116 = alloc(512 * 256);
    _Float16* w216 = alloc(256 * 512);
    _Float16* h16  = alloc(8192ull * 256);
    _Float16* q16  = alloc(8192ull * 256);
    _Float16* k16  = alloc(8192ull * 256);  // k16,v16 contiguous -> reused as ff
    _Float16* v16  = alloc(8192ull * 256);
    float* xres = (float*)(hb + off);       // 8192*256 f32 residual (x + attn)
    _Float16* ao16 = h16;                   // reuse: h dead after QKV
    _Float16* g16  = q16;                   // reuse: q dead after attention
    _Float16* ff16 = k16;                   // reuse: k,v dead after attention (8192x512)

    // 1) weights -> f16
    cvt_f16<<<256, 256, 0, stream>>>(Wq, wq16, 256 * 256);
    cvt_f16<<<256, 256, 0, stream>>>(Wk, wk16, 256 * 256);
    cvt_f16<<<256, 256, 0, stream>>>(Wv, wv16, 256 * 256);
    cvt_f16<<<256, 256, 0, stream>>>(Wo, wo16, 256 * 256);
    cvt_f16<<<512, 256, 0, stream>>>(W1, w116, 512 * 256);
    cvt_f16<<<512, 256, 0, stream>>>(W2, w216, 256 * 512);

    // 2) LN + PE + transpose -> h16 (B*K, T, D)
    ln_kernel<true><<<1024, 256, 0, stream>>>(x, ln_t_g, ln_t_b, h16);

    // 3) Q, K, V projections (f16 out)
    gemm_tile<0, 256, 256><<<dim3(128, 4), 256, 0, stream>>>(h16, wq16, bq, q16, nullptr);
    gemm_tile<0, 256, 256><<<dim3(128, 4), 256, 0, stream>>>(h16, wk16, bk, k16, nullptr);
    gemm_tile<0, 256, 256><<<dim3(128, 4), 256, 0, stream>>>(h16, wv16, bv, v16, nullptr);

    // 4) windowed attention -> ao16
    attn_kernel<<<512, 256, 0, stream>>>(q16, k16, v16, ao16);

    // 5) O-projection + residual, back to (B,T,K,D) -> xres (f32)
    gemm_tile<2, 256, 256><<<dim3(128, 4), 256, 0, stream>>>(ao16, wo16, bo, xres, x);

    // 6) final LN -> g16
    ln_kernel<false><<<1024, 256, 0, stream>>>(xres, ln_f_g, ln_f_b, g16);

    // 7) FFN1 (ReLU, f16) and FFN2 (+residual -> f32 output)
    gemm_tile<1, 256, 512><<<dim3(128, 8), 256, 0, stream>>>(g16, w116, b1, ff16, nullptr);
    gemm_tile<3, 512, 256><<<dim3(128, 4), 256, 0, stream>>>(ff16, w216, b2, d_out, xres);
}
